// TransformerConvNet_85315230367963
// MI455X (gfx1250) — matrix operand order repaired and checked by reading, not compiled, not run on hardware
//
#include <hip/hip_runtime.h>
#include <hip/hip_bf16.h>

typedef __attribute__((ext_vector_type(16))) _Float16 v16h;
typedef __attribute__((ext_vector_type(8)))  _Float16 v8h;
typedef __attribute__((ext_vector_type(8)))  float    v8f;

#define B_   32
#define N_   128
#define H_   128
#define HD_  256
#define DH_  64

// ---------------------------------------------------------------------------
// WMMA fragment helpers (CDNA5 16x16x32 f16, wave32)
// A (16x32, MxK) layout: lane l: M=l&15; halves i<8 -> K=k0+i+((l&16)?8:0),
//                        halves i>=8 -> K=k0+16+(i-8)+((l&16)?8:0)
// B (32x16, KxN) layout: same K mapping as A; lane l: N=l&15
// C/D (16x16 f32):       lane l: N=l&15; vgpr r -> M=r+((l&16)?8:0)
// ---------------------------------------------------------------------------
static __device__ __forceinline__ v16h frag_A(const _Float16* p, int row0, int ld, int k0) {
  const int l   = threadIdx.x & 31;
  const int m   = row0 + (l & 15);
  const int khi = (l & 16) ? 8 : 0;
  const _Float16* q = p + m * ld + k0 + khi;
  v8h lo = *(const v8h*)(q);
  v8h hi = *(const v8h*)(q + 16);
  return __builtin_shufflevector(lo, hi, 0,1,2,3,4,5,6,7,8,9,10,11,12,13,14,15);
}

// B fragment from column-major (transposed) storage: element(k,n) = p[n*ld + k]
static __device__ __forceinline__ v16h frag_Bt(const _Float16* p, int k0, int n0, int ld) {
  const int l   = threadIdx.x & 31;
  const int n   = n0 + (l & 15);
  const int khi = (l & 16) ? 8 : 0;
  const _Float16* q = p + n * ld + k0 + khi;
  v8h lo = *(const v8h*)(q);
  v8h hi = *(const v8h*)(q + 16);
  return __builtin_shufflevector(lo, hi, 0,1,2,3,4,5,6,7,8,9,10,11,12,13,14,15);
}

static __device__ __forceinline__ v8f wmma16(v16h a, v16h b, v8f c) {
  return __builtin_amdgcn_wmma_f32_16x16x32_f16(false, a, false, b, (short)0, c, false, false);
}

static __device__ __forceinline__ void store_C(float* c, const v8f& acc, int row0, int col0, int ld) {
  const int l  = threadIdx.x & 31;
  const int n  = col0 + (l & 15);
  const int m0 = row0 + ((l & 16) ? 8 : 0);
#pragma unroll
  for (int r = 0; r < 8; ++r) c[(m0 + r) * ld + n] = acc[r];
}

// ---------------------------------------------------------------------------
// base[b,s,:] = concat(node_feat, emb[entity]) @ W1[:17] + b1   (rank-17 part)
// ---------------------------------------------------------------------------
__global__ __launch_bounds__(128) void k_base(const float* __restrict__ nf,
                                              const int* __restrict__ et,
                                              const float* __restrict__ emb,
                                              const float* __restrict__ W1,
                                              const float* __restrict__ b1,
                                              float* __restrict__ base) {
  const int row = blockIdx.x, c = threadIdx.x;
  float acc = b1[c];
  const float* f = nf + row * 9;
#pragma unroll
  for (int j = 0; j < 9; ++j) acc += f[j] * W1[j * H_ + c];
  const float* e = emb + et[row] * 8;
#pragma unroll
  for (int j = 0; j < 8; ++j) acc += e[j] * W1[(9 + j) * H_ + c];
  base[(size_t)row * H_ + c] = acc;
}

// convert f32 (K x N) -> f16 transposed (N x K)
__global__ void k_cvt_t(const float* __restrict__ in, _Float16* __restrict__ out, int K, int N) {
  int i = blockIdx.x * 256 + threadIdx.x;
  if (i < K * N) {
    int k = i / N, n = i % N;
    out[(size_t)n * K + k] = (_Float16)in[i];
  }
}

// ---------------------------------------------------------------------------
// Fused embed-conv: per block (b, 16 t's). Loop s in chunks of 8 (128 rows):
//   h1 = LN1(relu(base[s] + adjv[s,t]*W1[17]))  -> f16 A in LDS
//   C  = A @ Wh (WMMA)                          -> f32 in LDS
//   x[t] += mask[s,t] * LN2(relu(C + bh))
// ---------------------------------------------------------------------------
__global__ __launch_bounds__(256) void k_embed(
    const float* __restrict__ adj, const float* __restrict__ base,
    const float* __restrict__ W1, const float* __restrict__ g1, const float* __restrict__ be1,
    const _Float16* __restrict__ WhT, const float* __restrict__ bh,
    const float* __restrict__ gh, const float* __restrict__ beh,
    float* __restrict__ x1, _Float16* __restrict__ x1h) {
  __shared__ __align__(16) _Float16 sWhT[H_ * H_];   // 32 KB  [n][k]
  __shared__ __align__(16) _Float16 sA[128 * H_];    // 32 KB  rows = sl*16+tl
  __shared__ __align__(16) float    sC[128 * H_];    // 64 KB
  __shared__ float sAcc[16 * H_];                    // 8 KB
  __shared__ float sStats[128 * 2];
  __shared__ float sAdj[128];
  __shared__ float sMask[128];

  const int tid = threadIdx.x;
  const int b   = blockIdx.x >> 3;
  const int t0  = (blockIdx.x & 7) * 16;
  const int wid = tid >> 5;

  for (int i = tid; i < H_ * H_; i += 256) sWhT[i] = WhT[i];
  for (int i = tid; i < 16 * H_; i += 256) sAcc[i] = 0.f;
  __syncthreads();

  const float* w17 = W1 + 17 * H_;

  for (int s0 = 0; s0 < N_; s0 += 8) {
    if (tid < 128) {
      int sl = tid >> 4, tl = tid & 15;
      float a = adj[((size_t)b * N_ + (s0 + sl)) * N_ + t0 + tl];
      float m = (a > 0.f && a < 1.f) ? 1.f : 0.f;
      sAdj[tid] = a * m;
      sMask[tid] = m;
    }
    __syncthreads();

    // pass1: raw relu rows -> sC, stats per row (thread pair via shfl_xor)
    {
      int r = tid >> 1, c0 = (tid & 1) * 64;
      float av = sAdj[r];
      const float* brow = base + ((size_t)b * N_ + s0 + (r >> 4)) * H_;
      float sum = 0.f, sq = 0.f;
      for (int c = 0; c < 64; ++c) {
        float v = fmaxf(brow[c0 + c] + av * w17[c0 + c], 0.f);
        sC[r * H_ + c0 + c] = v;
        sum += v; sq += v * v;
      }
      sum += __shfl_xor(sum, 1);
      sq  += __shfl_xor(sq, 1);
      if ((tid & 1) == 0) {
        float mu  = sum * (1.f / 128.f);
        float var = sq * (1.f / 128.f) - mu * mu;
        sStats[r * 2] = mu;
        sStats[r * 2 + 1] = rsqrtf(var + 1e-5f);
      }
    }
    __syncthreads();
    // pass2: LN1 -> f16 A
    for (int i = tid; i < 128 * H_; i += 256) {
      int r = i >> 7, c = i & 127;
      sA[i] = (_Float16)((sC[i] - sStats[r * 2]) * sStats[r * 2 + 1] * g1[c] + be1[c]);
    }
    __syncthreads();
    // GEMM: 128x128x128; wave wid owns one 16-wide N tile
    {
      const int n0 = wid * 16;
#pragma unroll
      for (int mt = 0; mt < 8; ++mt) {
        v8f acc = {};
#pragma unroll
        for (int ks = 0; ks < 4; ++ks) {
          v16h a  = frag_A(sA, mt * 16, H_, ks * 32);
          v16h bb = frag_Bt(sWhT, ks * 32, n0, H_);
          acc = wmma16(a, bb, acc);
        }
        store_C(sC, acc, mt * 16, n0, H_);
      }
    }
    __syncthreads();
    // pass3: stats of relu(C + bh)
    {
      int r = tid >> 1, c0 = (tid & 1) * 64;
      float sum = 0.f, sq = 0.f;
      for (int c = 0; c < 64; ++c) {
        float v = fmaxf(sC[r * H_ + c0 + c] + bh[c0 + c], 0.f);
        sum += v; sq += v * v;
      }
      sum += __shfl_xor(sum, 1);
      sq  += __shfl_xor(sq, 1);
      if ((tid & 1) == 0) {
        float mu  = sum * (1.f / 128.f);
        float var = sq * (1.f / 128.f) - mu * mu;
        sStats[r * 2] = mu;
        sStats[r * 2 + 1] = rsqrtf(var + 1e-5f);
      }
    }
    __syncthreads();
    // pass4: masked accumulate into sAcc (thread owns (t, 8-col slice))
    {
      int tl = tid >> 4, c0 = (tid & 15) * 8;
#pragma unroll
      for (int sl = 0; sl < 8; ++sl) {
        int r = sl * 16 + tl;
        if (sMask[r] > 0.f) {
          float mu = sStats[r * 2], rs = sStats[r * 2 + 1];
#pragma unroll
          for (int j = 0; j < 8; ++j) {
            int c = c0 + j;
            float v = fmaxf(sC[r * H_ + c] + bh[c], 0.f);
            sAcc[tl * H_ + c] += (v - mu) * rs * gh[c] + beh[c];
          }
        }
      }
    }
    __syncthreads();
  }
  for (int i = tid; i < 16 * H_; i += 256) {
    int tl = i >> 7, c = i & 127;
    float v = sAcc[i];
    size_t g = ((size_t)b * N_ + t0 + tl) * H_ + c;
    x1[g] = v;
    x1h[g] = (_Float16)v;
  }
}

// ---------------------------------------------------------------------------
// Generic projection GEMM: Y(M x 256) = X(M x C, f16) @ W(C x 256) + bias
// WT is f16 transposed (256 x C). M-tile 64 per block, 8 waves.
// ---------------------------------------------------------------------------
__global__ __launch_bounds__(256) void k_gemm(const _Float16* __restrict__ X,
                                              const _Float16* __restrict__ WT,
                                              const float* __restrict__ bias,
                                              float* __restrict__ Y, int C) {
  __shared__ __align__(16) _Float16 sX[64 * 64];    // 8 KB  [r][kk]
  __shared__ __align__(16) _Float16 sW[256 * 64];   // 32 KB [n][kk]
  const int tid = threadIdx.x, wid = tid >> 5;
  const int m0 = blockIdx.x * 64;
  v8f acc[4][2] = {};

  for (int kc = 0; kc < C; kc += 64) {
    for (int i = tid; i < 64 * 64; i += 256) {
      int r = i >> 6, k = i & 63;
      sX[i] = X[(size_t)(m0 + r) * C + kc + k];
    }
    for (int i = tid; i < 256 * 64; i += 256) {
      int n = i >> 6, k = i & 63;
      sW[i] = WT[(size_t)n * C + kc + k];
    }
    __syncthreads();
#pragma unroll
    for (int mt = 0; mt < 4; ++mt)
#pragma unroll
      for (int nn = 0; nn < 2; ++nn)
#pragma unroll
        for (int ks = 0; ks < 2; ++ks) {
          v16h a  = frag_A(sX, mt * 16, 64, ks * 32);
          v16h bb = frag_Bt(sW, ks * 32, wid * 32 + nn * 16, 64);
          acc[mt][nn] = wmma16(a, bb, acc[mt][nn]);
        }
    __syncthreads();
  }
  const int l = tid & 31;
#pragma unroll
  for (int mt = 0; mt < 4; ++mt)
#pragma unroll
    for (int nn = 0; nn < 2; ++nn) {
      int n = wid * 32 + nn * 16 + (l & 15);
      int m = m0 + mt * 16 + ((l & 16) ? 8 : 0);
#pragma unroll
      for (int r = 0; r < 8; ++r)
        Y[(size_t)(m + r) * HD_ + n] = acc[mt][nn][r] + bias[n];
    }
}

// ---------------------------------------------------------------------------
// Attention per (b, head): logits = K Q^T (WMMA) + adjv*qe, masked softmax
// over s (f32), out = alpha^T V (WMMA) + (sum_s alpha*adjv) * we
// ---------------------------------------------------------------------------
__global__ __launch_bounds__(256) void k_attn(const float* __restrict__ Q,
                                              const float* __restrict__ Kp,
                                              const float* __restrict__ V,
                                              const float* __restrict__ adj,
                                              const float* __restrict__ We,
                                              float* __restrict__ O) {
  __shared__ __align__(16) _Float16 sK[128 * 64];    // 16 KB [s][d]
  __shared__ __align__(16) _Float16 sQ[128 * 64];    // 16 KB [t][d]
  __shared__ __align__(16) _Float16 sVt[64 * 128];   // 16 KB [d][s]
  __shared__ __align__(16) float    sL[128 * 128];   // 64 KB [s][t]
  __shared__ __align__(16) _Float16 sAl[128 * 128];  // 32 KB [t][s]
  __shared__ float sQe[128];
  __shared__ float sEo[128];

  const int tid = threadIdx.x, wid = tid >> 5;
  const int b = blockIdx.x >> 2, h = blockIdx.x & 3;

  for (int i = tid; i < 128 * 64; i += 256) {
    int r = i >> 6, d = i & 63;
    size_t g = ((size_t)b * N_ + r) * HD_ + h * DH_ + d;
    sK[i] = (_Float16)Kp[g];
    sQ[i] = (_Float16)Q[g];
    sVt[d * 128 + r] = (_Float16)V[g];
  }
  __syncthreads();
  // qe[t] = q[t,:] . we
  {
    int t = tid >> 1, c0 = (tid & 1) * 32;
    float s = 0.f;
    for (int j = 0; j < 32; ++j) s += (float)sQ[t * 64 + c0 + j] * We[h * DH_ + c0 + j];
    s += __shfl_xor(s, 1);
    if ((tid & 1) == 0) sQe[t] = s;
  }
  __syncthreads();
  // GEMM1: logits[s][t]; wave wid owns N-tile (t) wid
  {
    const int n0 = wid * 16;
#pragma unroll
    for (int mt = 0; mt < 8; ++mt) {
      v8f acc = {};
#pragma unroll
      for (int ks = 0; ks < 2; ++ks) {
        v16h a  = frag_A(sK, mt * 16, 64, ks * 32);
        v16h bb = frag_Bt(sQ, ks * 32, n0, 64);  // B(k=d,n=t) = sQ[t*64+d]
        acc = wmma16(a, bb, acc);
      }
      store_C(sL, acc, mt * 16, n0, 128);
    }
  }
  __syncthreads();
  // masked softmax over s per column t (thread pair handles 64 s each)
  {
    const float scale = 0.125f;  // 1/sqrt(64)
    int t = tid >> 1, s0 = (tid & 1) * 64;
    float qe = sQe[t];
    float mx = -3.0e38f;
    for (int s = s0; s < s0 + 64; ++s) {
      float a  = adj[((size_t)b * N_ + s) * N_ + t];
      float m  = (a > 0.f && a < 1.f) ? 1.f : 0.f;
      float av = a * m;
      float lg = (m > 0.f) ? (sL[s * 128 + t] + av * qe) * scale : -1.0e30f;
      sL[s * 128 + t] = lg;
      sAl[t * 128 + s] = (_Float16)av;  // stash adjv (in (0,1))
      mx = fmaxf(mx, lg);
    }
    mx = fmaxf(mx, __shfl_xor(mx, 1));
    float den = 0.f, eo = 0.f;
    for (int s = s0; s < s0 + 64; ++s) {
      float av = (float)sAl[t * 128 + s];
      float w  = (av > 0.f) ? __expf(sL[s * 128 + t] - mx) : 0.f;
      sAl[t * 128 + s] = (_Float16)w;
      den += w;
      eo += w * av;
    }
    den += __shfl_xor(den, 1);
    eo  += __shfl_xor(eo, 1);
    float inv = 1.f / fmaxf(den, 1e-16f);
    for (int s = s0; s < s0 + 64; ++s)
      sAl[t * 128 + s] = (_Float16)((float)sAl[t * 128 + s] * inv);
    if ((tid & 1) == 0) sEo[t] = eo * inv;
  }
  __syncthreads();
  // GEMM2: out[t][d] = alpha[t][s] V[s][d]; wave wid owns M-tile (t) wid
  {
    const int mt = wid;
    const int l = tid & 31;
#pragma unroll
    for (int nt = 0; nt < 4; ++nt) {
      v8f acc = {};
#pragma unroll
      for (int ks = 0; ks < 4; ++ks) {
        v16h a  = frag_A(sAl, mt * 16, 128, ks * 32);
        v16h bb = frag_Bt(sVt, ks * 32, nt * 16, 128);  // B(k=s,n=d) = sVt[d*128+s]
        acc = wmma16(a, bb, acc);
      }
      int d = nt * 16 + (l & 15);
      int m = mt * 16 + ((l & 16) ? 8 : 0);
      float wed = We[h * DH_ + d];
#pragma unroll
      for (int r = 0; r < 8; ++r) {
        int t = m + r;
        O[((size_t)b * N_ + t) * HD_ + h * DH_ + d] = acc[r] + sEo[t] * wed;
      }
    }
  }
}

__global__ void k_combine(const float* __restrict__ Oin, const float* __restrict__ S,
                          float* __restrict__ Xo, _Float16* __restrict__ Xh, int n) {
  int i = blockIdx.x * 256 + threadIdx.x;
  if (i < n) {
    float v = fmaxf(Oin[i] + S[i], 0.f);
    Xo[i] = v;
    Xh[i] = (_Float16)v;
  }
}

__global__ void k_gather(const float* __restrict__ X, const int* __restrict__ agent,
                         float* __restrict__ out) {
  int b = blockIdx.x, c = threadIdx.x;
  out[(size_t)b * HD_ + c] = X[((size_t)b * N_ + agent[b]) * HD_ + c];
}

// ---------------------------------------------------------------------------
extern "C" void kernel_launch(void* const* d_in, const int* in_sizes, int n_in,
                              void* d_out, int out_size, void* d_ws, size_t ws_size,
                              hipStream_t stream) {
  (void)in_sizes; (void)n_in; (void)out_size; (void)ws_size;
  const float* node_feat = (const float*)d_in[0];
  const int*   entity    = (const int*)d_in[1];
  const float* adj       = (const float*)d_in[2];
  const int*   agent     = (const int*)d_in[3];
  const float* emb       = (const float*)d_in[4];
  const float* W1  = (const float*)d_in[5];
  const float* b1  = (const float*)d_in[6];
  const float* g1  = (const float*)d_in[7];
  const float* be1 = (const float*)d_in[8];
  const float* Wh  = (const float*)d_in[9];
  const float* bh  = (const float*)d_in[10];
  const float* gh  = (const float*)d_in[11];
  const float* beh = (const float*)d_in[12];
  const float* Wq1 = (const float*)d_in[13]; const float* bq1 = (const float*)d_in[14];
  const float* Wk1 = (const float*)d_in[15]; const float* bk1 = (const float*)d_in[16];
  const float* Wv1 = (const float*)d_in[17]; const float* bv1 = (const float*)d_in[18];
  const float* We1 = (const float*)d_in[19];
  const float* Ws1 = (const float*)d_in[20]; const float* bs1 = (const float*)d_in[21];
  const float* Wq2 = (const float*)d_in[22]; const float* bq2 = (const float*)d_in[23];
  const float* Wk2 = (const float*)d_in[24]; const float* bk2 = (const float*)d_in[25];
  const float* Wv2 = (const float*)d_in[26]; const float* bv2 = (const float*)d_in[27];
  const float* We2 = (const float*)d_in[28];
  const float* Ws2 = (const float*)d_in[29]; const float* bs2 = (const float*)d_in[30];

  char* ws = (char*)d_ws;
  size_t off = 0;
  auto alloc = [&](size_t bytes) -> void* {
    void* p = ws + off;
    off += (bytes + 255) & ~(size_t)255;
    return p;
  };
  const size_t RN = (size_t)B_ * N_;           // 4096 rows
  float*    base = (float*)alloc(RN * H_ * 4);
  float*    x1   = (float*)alloc(RN * H_ * 4);
  _Float16* x1h  = (_Float16*)alloc(RN * H_ * 2);
  _Float16* WhT  = (_Float16*)alloc((size_t)H_ * H_ * 2);
  _Float16* Wq1T = (_Float16*)alloc((size_t)H_ * HD_ * 2);
  _Float16* Wk1T = (_Float16*)alloc((size_t)H_ * HD_ * 2);
  _Float16* Wv1T = (_Float16*)alloc((size_t)H_ * HD_ * 2);
  _Float16* Ws1T = (_Float16*)alloc((size_t)H_ * HD_ * 2);
  _Float16* Wq2T = (_Float16*)alloc((size_t)HD_ * HD_ * 2);
  _Float16* Wk2T = (_Float16*)alloc((size_t)HD_ * HD_ * 2);
  _Float16* Wv2T = (_Float16*)alloc((size_t)HD_ * HD_ * 2);
  _Float16* Ws2T = (_Float16*)alloc((size_t)HD_ * HD_ * 2);
  float*    Qb   = (float*)alloc(RN * HD_ * 4);
  float*    Kb   = (float*)alloc(RN * HD_ * 4);
  float*    Vb   = (float*)alloc(RN * HD_ * 4);
  float*    Sb   = (float*)alloc(RN * HD_ * 4);
  float*    Ob   = (float*)alloc(RN * HD_ * 4);
  float*    x2   = (float*)alloc(RN * HD_ * 4);
  _Float16* x2h  = (_Float16*)alloc(RN * HD_ * 2);
  float*    x3   = (float*)alloc(RN * HD_ * 4);
  _Float16* x3h  = (_Float16*)alloc(RN * HD_ * 2);

  // --- prep ---
  k_base<<<RN, 128, 0, stream>>>(node_feat, entity, emb, W1, b1, base);
  k_cvt_t<<<(H_ * H_ + 255) / 256, 256, 0, stream>>>(Wh, WhT, H_, H_);
  k_cvt_t<<<(H_ * HD_ + 255) / 256, 256, 0, stream>>>(Wq1, Wq1T, H_, HD_);
  k_cvt_t<<<(H_ * HD_ + 255) / 256, 256, 0, stream>>>(Wk1, Wk1T, H_, HD_);
  k_cvt_t<<<(H_ * HD_ + 255) / 256, 256, 0, stream>>>(Wv1, Wv1T, H_, HD_);
  k_cvt_t<<<(H_ * HD_ + 255) / 256, 256, 0, stream>>>(Ws1, Ws1T, H_, HD_);
  k_cvt_t<<<(HD_ * HD_ + 255) / 256, 256, 0, stream>>>(Wq2, Wq2T, HD_, HD_);
  k_cvt_t<<<(HD_ * HD_ + 255) / 256, 256, 0, stream>>>(Wk2, Wk2T, HD_, HD_);
  k_cvt_t<<<(HD_ * HD_ + 255) / 256, 256, 0, stream>>>(Wv2, Wv2T, HD_, HD_);
  k_cvt_t<<<(HD_ * HD_ + 255) / 256, 256, 0, stream>>>(Ws2, Ws2T, HD_, HD_);

  // --- embed conv (fused, WMMA) ---
  k_embed<<<B_ * (N_ / 16), 256, 0, stream>>>(adj, base, W1, g1, be1, WhT, bh, gh, beh, x1, x1h);

  const int nElem = (int)(RN * HD_);
  const int gM = (int)(RN / 64);

  // --- tconv layer 1 ---
  k_gemm<<<gM, 256, 0, stream>>>(x1h, Wq1T, bq1, Qb, H_);
  k_gemm<<<gM, 256, 0, stream>>>(x1h, Wk1T, bk1, Kb, H_);
  k_gemm<<<gM, 256, 0, stream>>>(x1h, Wv1T, bv1, Vb, H_);
  k_gemm<<<gM, 256, 0, stream>>>(x1h, Ws1T, bs1, Sb, H_);
  k_attn<<<B_ * 4, 256, 0, stream>>>(Qb, Kb, Vb, adj, We1, Ob);
  k_combine<<<(nElem + 255) / 256, 256, 0, stream>>>(Ob, Sb, x2, x2h, nElem);

  // --- tconv layer 2 ---
  k_gemm<<<gM, 256, 0, stream>>>(x2h, Wq2T, bq2, Qb, HD_);
  k_gemm<<<gM, 256, 0, stream>>>(x2h, Wk2T, bk2, Kb, HD_);
  k_gemm<<<gM, 256, 0, stream>>>(x2h, Wv2T, bv2, Vb, HD_);
  k_gemm<<<gM, 256, 0, stream>>>(x2h, Ws2T, bs2, Sb, HD_);
  k_attn<<<B_ * 4, 256, 0, stream>>>(Qb, Kb, Vb, adj, We2, Ob);
  k_combine<<<(nElem + 255) / 256, 256, 0, stream>>>(Ob, Sb, x3, x3h, nElem);

  // --- gather agent rows ---
  k_gather<<<B_, 256, 0, stream>>>(x3, agent, (float*)d_out);
}
